// MultiHeadAttention_24094766530600
// MI455X (gfx1250) — matrix operand order repaired and checked by reading, not compiled, not run on hardware
//
#include <hip/hip_runtime.h>
#include <hip/hip_bf16.h>

// ---------------------------------------------------------------------------
// MHA (B=8, S=1024, D=1024, H=16, dh=64, flat "quirky" reshape, scale=0.5)
// f16 WMMA (v_wmma_f32_16x16x32_f16) everywhere, f32 softmax/bias/BN.
// TDM (tensor_load_to_lds) double-buffered V pipeline; global_prefetch (WGP).
// ---------------------------------------------------------------------------

typedef __attribute__((ext_vector_type(16))) _Float16 v16h;
typedef __attribute__((ext_vector_type(8)))  _Float16 v8h;
typedef __attribute__((ext_vector_type(8)))  float    v8f;
typedef unsigned int u32;
typedef __attribute__((ext_vector_type(4))) u32 u32x4;
typedef __attribute__((ext_vector_type(8))) int i32x8;
typedef __attribute__((ext_vector_type(4))) int i32x4;

#define DMODEL 1024
#define ROWS   8192      // B*S
#define NHEADS 128       // B*H after flat reshape
#define SEQ    1024      // per-head rows after flat reshape
#define DH     64
#define ATTN_SCALE 0.5f  // (dh // H)^-0.5 = 4^-0.5

// ---- WMMA helpers ---------------------------------------------------------

__device__ __forceinline__ v8f wmma_f16(v16h a, v16h b, v8f c) {
  return __builtin_amdgcn_wmma_f32_16x16x32_f16(
      /*neg_a=*/false, a, /*neg_b=*/false, b,
      /*c_mod=*/(short)0, c, /*reuse_a=*/false, /*reuse_b=*/false);
}

// A-matrix 16x32 fragment: lane L -> M=L%16, g=L/16, K = 16g + h (contiguous)
__device__ __forceinline__ v16h frag_a(const _Float16* p) {
  v8h lo = *(const v8h*)(p);
  v8h hi = *(const v8h*)(p + 8);
  v16h f;
#pragma unroll
  for (int j = 0; j < 8; ++j) { f[j] = lo[j]; f[j + 8] = hi[j]; }
  return f;
}

// B-matrix 32x16 fragment: lane L -> N=L%16, g=L/16, K = 16g + h (contiguous)
__device__ __forceinline__ v16h frag_b(const _Float16* p) {
  v8h lo = *(const v8h*)(p);
  v8h hi = *(const v8h*)(p + 8);
  v16h f;
#pragma unroll
  for (int j = 0; j < 8; ++j) { f[j] = lo[j]; f[j + 8] = hi[j]; }
  return f;
}

// ---- Tensor Data Mover: 2D tile (16-bit elements) global -> LDS -----------
// D# groups per CDNA5 ISA; tile == tensor region.
// 6-arg builtin: (uint32x4 g0, int32x8 g1, i32x4 g2, i32x4 g3, i32x8, i32 cpol)
__device__ __forceinline__ void tdm_load_2d_b16(u32 lds_off, const void* gaddr,
                                                u32 dim0, u32 dim1,
                                                u32 stride0) {
  unsigned long long ga = (unsigned long long)gaddr;
  u32x4 g0;
  g0[0] = 1u;                                 // count=1, is_restore=0
  g0[1] = lds_off;                            // lds_addr (bytes)
  g0[2] = (u32)(ga & 0xffffffffull);          // global_addr[31:0]
  g0[3] = (u32)((ga >> 32) & 0x1ffffffull) |  // global_addr[56:32]
          (2u << 30);                         // type = 2 ("image")
  i32x8 g1;
  g1[0] = (int)(1u << 16);                    // wg_mask=0, data_size=1 (2B)
  g1[1] = (int)((dim0 & 0xffffu) << 16);      // tensor_dim0[15:0] @bits63:48
  g1[2] = (int)(((dim0 >> 16) & 0xffffu) |    // tensor_dim0[31:16]
                ((dim1 & 0xffffu) << 16));    // tensor_dim1[15:0]
  g1[3] = (int)(((dim1 >> 16) & 0xffffu) |    // tensor_dim1[31:16]
                ((dim0 & 0xffffu) << 16));    // tile_dim0 = dim0
  g1[4] = (int)(dim1 & 0xffffu);              // tile_dim1 = dim1, tile_dim2=0
  g1[5] = (int)stride0;                       // tensor_dim0_stride[31:0]
  g1[6] = 0;                                  // stride0[47:32], stride1[15:0]
  g1[7] = 0;                                  // stride1[47:16]
  i32x4 z4;
  z4[0] = 0; z4[1] = 0; z4[2] = 0; z4[3] = 0;
  i32x8 z8;
#pragma unroll
  for (int j = 0; j < 8; ++j) z8[j] = 0;
  __builtin_amdgcn_tensor_load_to_lds(g0, g1, z4, z4, z8, 0);
}

// ---- GEMM: C[8192x1024] = A[8192x1024] @ W[1024x1024] (+bias)(+residual) ---
// Block tile 128(M) x 64(N); 4 waves, each computes a 32x64 strip: 8 WMMA/step
template <bool A_HALF, bool OUT_RES>
__global__ __launch_bounds__(128, 1) void gemm1024(
    const void* __restrict__ Asrc, const float* __restrict__ W,
    const float* __restrict__ bias, const float* __restrict__ resid,
    void* __restrict__ outv) {
  __shared__ _Float16 Asl[128 * 32];  // [m][k], k contiguous (8 KB)
  __shared__ _Float16 Bsl[64 * 32];   // transposed: [n][k] (4 KB)

  const int tid  = threadIdx.x;
  const int wave = tid >> 5;
  const int lane = tid & 31;
  const int li   = lane & 15;
  const int lg   = (lane >> 4) & 1;
  const int n0   = blockIdx.x * 64;
  const int m0   = blockIdx.y * 128;

  v8f acc[2][4];
#pragma unroll
  for (int ms = 0; ms < 2; ++ms)
#pragma unroll
    for (int t = 0; t < 4; ++t)
#pragma unroll
      for (int r = 0; r < 8; ++r) acc[ms][t][r] = 0.0f;

  for (int k0 = 0; k0 < DMODEL; k0 += 32) {
    __syncthreads();
    {  // A tile 128x32 -> LDS (f16); one row per thread
      const int row = tid;
      if (A_HALF) {
        const _Float16* src =
            (const _Float16*)Asrc + (size_t)(m0 + row) * DMODEL + k0;
        __builtin_prefetch(src + 32, 0, 3);
#pragma unroll
        for (int q8 = 0; q8 < 4; ++q8)
          *(v8h*)(Asl + row * 32 + q8 * 8) = *(const v8h*)(src + q8 * 8);
      } else {
        const float* src =
            (const float*)Asrc + (size_t)(m0 + row) * DMODEL + k0;
        __builtin_prefetch(src + 32, 0, 3);
#pragma unroll
        for (int q8 = 0; q8 < 4; ++q8) {
          v8h t;
#pragma unroll
          for (int j = 0; j < 8; ++j) t[j] = (_Float16)src[q8 * 8 + j];
          *(v8h*)(Asl + row * 32 + q8 * 8) = t;
        }
      }
    }
    {  // W tile 32x64 -> LDS transposed [n][k] (f16)
      const int k = tid >> 2, seg = (tid & 3) * 16;
      const float* src = W + (size_t)(k0 + k) * DMODEL + n0 + seg;
      __builtin_prefetch(src + 32 * DMODEL, 0, 3);
#pragma unroll
      for (int j = 0; j < 16; ++j)
        Bsl[(seg + j) * 32 + k] = (_Float16)src[j];
    }
    __syncthreads();

    v16h af0 = frag_a(Asl + (32 * wave + li) * 32 + 16 * lg);
    v16h af1 = frag_a(Asl + (32 * wave + 16 + li) * 32 + 16 * lg);
#pragma unroll
    for (int nt = 0; nt < 4; ++nt) {
      v16h bf = frag_b(Bsl + (nt * 16 + li) * 32 + 16 * lg);
      acc[0][nt] = wmma_f16(af0, bf, acc[0][nt]);
      acc[1][nt] = wmma_f16(af1, bf, acc[1][nt]);
    }
  }

#pragma unroll
  for (int ms = 0; ms < 2; ++ms) {
#pragma unroll
    for (int nt = 0; nt < 4; ++nt) {
      const int n = n0 + nt * 16 + li;
      const float bn = bias[n];
#pragma unroll
      for (int r = 0; r < 8; ++r) {
        const int m = m0 + 32 * wave + 16 * ms + r + 8 * lg;
        const size_t idx = (size_t)m * DMODEL + n;
        const float val = acc[ms][nt][r] + bn;
        if (OUT_RES)
          ((float*)outv)[idx] = resid[idx] + val;
        else
          ((_Float16*)outv)[idx] = (_Float16)val;
      }
    }
  }
}

// ---- Attention: one block per (head bh, 16-row query block qb) -------------
#define VSTAGE  128                   // keys per V pipeline stage
#define NSTAGES (SEQ / VSTAGE)        // 8
#define SM_S   0                      // float [16][1024]        65536 B
#define SM_P   65536                  // f16   [16][1024]        32768 B
#define SM_Q   (65536 + 32768)        // f16   [16][64]           2048 B
#define SM_V   (65536 + 32768 + 2048) // f16   2 x [128][64]     32768 B
#define SM_R   (65536 + 32768 + 2048 + 32768)  // float [16][8]*2  1024 B
#define ATTN_SMEM (SM_R + 1024)

__global__ __launch_bounds__(128, 1) void attn_kernel(
    const _Float16* __restrict__ qh, const _Float16* __restrict__ kh,
    const _Float16* __restrict__ vh, _Float16* __restrict__ ctxh,
    float* __restrict__ attn_out) {
  extern __shared__ char smem[];
  float*    Ssl   = (float*)(smem + SM_S);
  _Float16* Psl   = (_Float16*)(smem + SM_P);
  _Float16* Qsl   = (_Float16*)(smem + SM_Q);
  _Float16* Vbase = (_Float16*)(smem + SM_V);  // 2 buffers, [key][d] each
  float*    red   = (float*)(smem + SM_R);
  float*    red2  = red + 128;

  const int tid  = threadIdx.x;
  const int wave = tid >> 5;
  const int lane = tid & 31;
  const int li   = lane & 15;
  const int lg   = (lane >> 4) & 1;
  const int bh   = blockIdx.x >> 6;  // 0..127
  const int qb   = blockIdx.x & 63;  // 0..63

  const _Float16* qhead = qh + (size_t)bh * SEQ * DH;
  const _Float16* khead = kh + (size_t)bh * SEQ * DH;
  const _Float16* vhead = vh + (size_t)bh * SEQ * DH;

  // Prologue: TDM Q block [16x64] and V stage 0 [128x64] (wave 0 only).
  // V stage 0 streams in underneath the whole scores+softmax phase.
  if (wave == 0) {
    tdm_load_2d_b16((u32)(unsigned long long)(void*)Qsl,
                    qhead + (size_t)(qb * 16) * DH, DH, 16, DH);
    tdm_load_2d_b16((u32)(unsigned long long)(void*)Vbase, vhead, DH, VSTAGE,
                    DH);
    __builtin_amdgcn_s_wait_tensorcnt(1);  // Q resident (in-order completion)
  }
  __syncthreads();

  const v16h aq0 = frag_a(Qsl + li * 64 + 0 + 16 * lg);
  const v16h aq1 = frag_a(Qsl + li * 64 + 32 + 16 * lg);

  v8f zero;
#pragma unroll
  for (int r = 0; r < 8; ++r) zero[r] = 0.0f;

  // ---- scores: 64 key tiles of 16, round-robin over 4 waves ----
  for (int nt = wave; nt < 64; nt += 4) {
    const _Float16* krow = khead + (size_t)(nt * 16 + li) * DH;
    __builtin_prefetch(krow + 4 * 16 * DH, 0, 3);  // next tile, WGP scope
    v16h b0 = frag_b(krow + 16 * lg);
    v16h b1 = frag_b(krow + 32 + 16 * lg);
    v8f s = wmma_f16(aq0, b0, zero);
    s = wmma_f16(aq1, b1, s);
#pragma unroll
    for (int r = 0; r < 8; ++r)
      Ssl[(r + 8 * lg) * 1024 + nt * 16 + li] = s[r] * ATTN_SCALE;
  }
  __syncthreads();

  // ---- softmax over each of 16 rows (8 threads per row) ----
  {
    const int row = tid >> 3, s8 = tid & 7;
    float m = -3.4e38f;
    for (int c = s8; c < 1024; c += 8) m = fmaxf(m, Ssl[row * 1024 + c]);
    red[row * 8 + s8] = m;
    __syncthreads();
    float rm = red[row * 8];
#pragma unroll
    for (int j = 1; j < 8; ++j) rm = fmaxf(rm, red[row * 8 + j]);
    float sum = 0.0f;
    for (int c = s8; c < 1024; c += 8) {
      const float e = __expf(Ssl[row * 1024 + c] - rm);
      Ssl[row * 1024 + c] = e;
      sum += e;
    }
    red2[row * 8 + s8] = sum;
    __syncthreads();
    float tot = 0.0f;
#pragma unroll
    for (int j = 0; j < 8; ++j) tot += red2[row * 8 + j];
    const float inv = 1.0f / tot;
    const size_t aoff = ((size_t)bh * 1024 + qb * 16 + row) * 1024;
    for (int c = s8; c < 1024; c += 8) {
      const float p = Ssl[row * 1024 + c] * inv;
      attn_out[aoff + c] = p;
      Psl[row * 1024 + c] = (_Float16)p;
    }
  }
  __syncthreads();

  // ---- context: C[16x64] = P[16x1024] @ V[1024x64] ----
  // Double-buffered TDM pipeline: stage s+1 streams while stage s computes.
  v8f cacc;
#pragma unroll
  for (int r = 0; r < 8; ++r) cacc[r] = 0.0f;

  for (int s = 0; s < NSTAGES; ++s) {
    if (wave == 0) {
      if (s + 1 < NSTAGES) {
        tdm_load_2d_b16(
            (u32)(unsigned long long)(void*)(Vbase + ((s + 1) & 1) * VSTAGE * DH),
            vhead + (size_t)(s + 1) * VSTAGE * DH, DH, VSTAGE, DH);
        __builtin_amdgcn_s_wait_tensorcnt(1);  // stage s resident
      } else {
        __builtin_amdgcn_s_wait_tensorcnt(0);
      }
    }
    __syncthreads();
    const _Float16* Vb = Vbase + (s & 1) * VSTAGE * DH;
#pragma unroll
    for (int kt = 0; kt < 4; ++kt) {  // 4 x 32-key WMMA per stage
      const int kb = s * 4 + kt;
      v16h ap = frag_a(Psl + li * 1024 + kb * 32 + 16 * lg);
      v16h bv;  // B[K=key][N=d]: strided gather from untransposed V tile
#pragma unroll
      for (int h = 0; h < 16; ++h)
        bv[h] = Vb[(kt * 32 + 16 * lg + h) * DH + (wave * 16 + li)];
      cacc = wmma_f16(ap, bv, cacc);
    }
    __syncthreads();
  }

#pragma unroll
  for (int r = 0; r < 8; ++r) {
    const int m = qb * 16 + r + 8 * lg;
    const int d = wave * 16 + li;
    ctxh[(size_t)bh * SEQ * DH + (size_t)m * DH + d] = (_Float16)cacc[r];
  }
}

// ---- BatchNorm (training-mode batch stats, deterministic, no atomics) ------

#define BN_CHUNKS 32
#define BN_CROWS  (ROWS / BN_CHUNKS)  // 256 rows per chunk

__global__ __launch_bounds__(256) void bn_stats_partial(
    const float* __restrict__ x, float* __restrict__ psum,
    float* __restrict__ psum2) {
  const int d = blockIdx.x * blockDim.x + threadIdx.x;  // channel 0..1023
  const int chunk = blockIdx.y;
  float s = 0.0f, s2 = 0.0f;
  const int r0 = chunk * BN_CROWS;
  for (int r = r0; r < r0 + BN_CROWS; ++r) {
    const float v = x[(size_t)r * DMODEL + d];
    s += v;
    s2 += v * v;
  }
  psum[(size_t)chunk * DMODEL + d] = s;
  psum2[(size_t)chunk * DMODEL + d] = s2;
}

__global__ __launch_bounds__(256) void bn_stats_final(
    const float* __restrict__ psum, const float* __restrict__ psum2,
    float* __restrict__ mean, float* __restrict__ invstd) {
  const int d = blockIdx.x * blockDim.x + threadIdx.x;
  float s = 0.0f, s2 = 0.0f;
#pragma unroll
  for (int c = 0; c < BN_CHUNKS; ++c) {
    s += psum[(size_t)c * DMODEL + d];
    s2 += psum2[(size_t)c * DMODEL + d];
  }
  const float m = s * (1.0f / (float)ROWS);
  const float var = s2 * (1.0f / (float)ROWS) - m * m;
  mean[d] = m;
  invstd[d] = rsqrtf(var + 1e-5f);
}

__global__ __launch_bounds__(256) void bn_apply(
    const float* __restrict__ x, const float* __restrict__ mean,
    const float* __restrict__ invstd, const float* __restrict__ gamma,
    const float* __restrict__ beta, float* __restrict__ out) {
  const size_t idx = (size_t)blockIdx.x * blockDim.x + threadIdx.x;
  const int d = (int)(idx & (DMODEL - 1));
  out[idx] = (x[idx] - mean[d]) * invstd[d] * gamma[d] + beta[d];
}

// ---------------------------------------------------------------------------

extern "C" void kernel_launch(void* const* d_in, const int* in_sizes, int n_in,
                              void* d_out, int out_size, void* d_ws,
                              size_t ws_size, hipStream_t stream) {
  (void)in_sizes; (void)n_in; (void)out_size; (void)ws_size;

  const float* key   = (const float*)d_in[0];
  const float* value = (const float*)d_in[1];
  const float* query = (const float*)d_in[2];
  const float* Wk = (const float*)d_in[3];
  const float* bk = (const float*)d_in[4];
  const float* Wv = (const float*)d_in[5];
  const float* bv = (const float*)d_in[6];
  const float* Wq = (const float*)d_in[7];
  const float* bq = (const float*)d_in[8];
  const float* Wf = (const float*)d_in[9];
  const float* bf = (const float*)d_in[10];
  const float* gamma = (const float*)d_in[11];
  const float* beta  = (const float*)d_in[12];

  // Workspace layout (~96.6 MB total)
  const size_t NELEM = (size_t)ROWS * DMODEL;  // 8M
  char* ws = (char*)d_ws;
  _Float16* qh   = (_Float16*)ws;          // 16 MB
  _Float16* kh   = qh + NELEM;             // 16 MB
  _Float16* vh   = kh + NELEM;             // 16 MB
  _Float16* ctxh = vh + NELEM;             // 16 MB
  float* xbuf    = (float*)(ctxh + NELEM); // 32 MB
  float* mean    = xbuf + NELEM;           // 4 KB
  float* invstd  = mean + DMODEL;          // 4 KB
  float* psum    = invstd + DMODEL;        // 128 KB
  float* psum2   = psum + BN_CHUNKS * DMODEL;  // 128 KB

  float* out_main = (float*)d_out;          // [8192 x 1024]
  float* out_attn = out_main + NELEM;       // [128 x 1024 x 1024]

  // Allow >64KB dynamic LDS for the attention kernel.
  (void)hipFuncSetAttribute((const void*)attn_kernel,
                            hipFuncAttributeMaxDynamicSharedMemorySize,
                            ATTN_SMEM);

  const dim3 gGemm(DMODEL / 64, ROWS / 128);  // 16 x 64
  const dim3 bGemm(128);

  // q/k/v projections (f32 in -> f16 out)
  gemm1024<false, false><<<gGemm, bGemm, 0, stream>>>(query, Wq, bq, nullptr, qh);
  gemm1024<false, false><<<gGemm, bGemm, 0, stream>>>(key,   Wk, bk, nullptr, kh);
  gemm1024<false, false><<<gGemm, bGemm, 0, stream>>>(value, Wv, bv, nullptr, vh);

  // attention: 128 heads x 64 query blocks
  attn_kernel<<<dim3(NHEADS * 64), dim3(128), ATTN_SMEM, stream>>>(
      qh, kh, vh, ctxh, out_attn);

  // x = residual + context @ Wf + bf (f32)
  gemm1024<true, true><<<gGemm, bGemm, 0, stream>>>(ctxh, Wf, bf, query, xbuf);

  // batchnorm over (B,S) per channel: partial sums -> finalize -> apply
  bn_stats_partial<<<dim3(DMODEL / 256, BN_CHUNKS), dim3(256), 0, stream>>>(
      xbuf, psum, psum2);
  bn_stats_final<<<dim3(DMODEL / 256), dim3(256), 0, stream>>>(psum, psum2,
                                                               mean, invstd);
  bn_apply<<<dim3((unsigned)(NELEM / 256)), dim3(256), 0, stream>>>(
      xbuf, mean, invstd, gamma, beta, out_main);
}
